// RelativeMultiHeadAttention_76501957476428
// MI455X (gfx1250) — matrix operand order repaired and checked by reading, not compiled, not run on hardware
//
#include <hip/hip_runtime.h>

// ---------------------------------------------------------------------------
// RelativeMultiHeadAttention (Transformer-XL style) for MI455X / gfx1250.
// All GEMM stages run on v_wmma_f32_16x16x32_bf16 (wave32 WMMA); the attn@V
// stage stages V k-slabs into LDS with global_load_async_to_lds_b128
// (double-buffered, s_wait_asynccnt pipelined).
// B=16, T=512, D=512, H=16, DH=32, P=2T-1=1023.
// ---------------------------------------------------------------------------

static constexpr int nB  = 16;
static constexpr int nT  = 512;
static constexpr int nD  = 512;
static constexpr int nH  = 16;
static constexpr int nDH = 32;
static constexpr int nP  = 1023;

typedef __attribute__((ext_vector_type(16))) __bf16   v16bf;
typedef __attribute__((ext_vector_type(8)))  float    v8f;
typedef __attribute__((ext_vector_type(4)))  unsigned uint4x;
typedef __attribute__((ext_vector_type(4)))  float    float4x;

union AFrag { v16bf v; unsigned u[8]; uint4x q2[2]; };

__device__ __forceinline__ v8f wmma_bf16(v16bf a, v16bf b, v8f c) {
  return __builtin_amdgcn_wmma_f32_16x16x32_bf16(
      /*neg_a=*/false, a, /*neg_b=*/false, b,
      /*c_mod=*/(short)0, c, /*reuse_a=*/false, /*reuse_b=*/false);
}

// Convert 8 fp32 -> bf16 into elements [o, o+7] of a v16bf fragment.
__device__ __forceinline__ void pack8(v16bf& d, int o, float4x a, float4x b) {
  d[o + 0] = (__bf16)a[0]; d[o + 1] = (__bf16)a[1];
  d[o + 2] = (__bf16)a[2]; d[o + 3] = (__bf16)a[3];
  d[o + 4] = (__bf16)b[0]; d[o + 5] = (__bf16)b[1];
  d[o + 6] = (__bf16)b[2]; d[o + 7] = (__bf16)b[3];
}

// ---------------------------------------------------------------------------
// Kernel 1: bpp -> conv(1->16,3x3) -> relu -> conv(16->1,3x3) = bpp_bias[B,T,T]
// ---------------------------------------------------------------------------
__global__ __launch_bounds__(256) void conv_bias_kernel(
    const float* __restrict__ bpp, const float* __restrict__ c1w,
    const float* __restrict__ c1b, const float* __restrict__ c2w,
    const float* __restrict__ c2b, float* __restrict__ out) {
  __shared__ float in_t[20 * 20];
  __shared__ float h1[16 * 18 * 18];
  const int x0 = blockIdx.x * 16, y0 = blockIdx.y * 16, b = blockIdx.z;
  const int tid = threadIdx.y * 16 + threadIdx.x;
  const float* src = bpp + (size_t)b * nT * nT;

  for (int idx = tid; idx < 400; idx += 256) {
    int ly = idx / 20, lx = idx % 20;
    int gy = y0 - 2 + ly, gx = x0 - 2 + lx;
    float v = 0.f;
    if (gy >= 0 && gy < nT && gx >= 0 && gx < nT) v = src[(size_t)gy * nT + gx];
    in_t[idx] = v;
  }
  __syncthreads();

  for (int idx = tid; idx < 16 * 18 * 18; idx += 256) {
    int ch = idx / 324, rem = idx % 324;
    int y = rem / 18, x = rem % 18;
    float acc = c1b[ch];
#pragma unroll
    for (int dy = 0; dy < 3; ++dy)
#pragma unroll
      for (int dx = 0; dx < 3; ++dx)
        acc += in_t[(y + dy) * 20 + (x + dx)] * c1w[ch * 9 + dy * 3 + dx];
    h1[idx] = fmaxf(acc, 0.f);
  }
  __syncthreads();

  const int tx = threadIdx.x, ty = threadIdx.y;
  float acc = c2b[0];
  for (int ch = 0; ch < 16; ++ch)
#pragma unroll
    for (int dy = 0; dy < 3; ++dy)
#pragma unroll
      for (int dx = 0; dx < 3; ++dx)
        acc += h1[ch * 324 + (ty + dy) * 18 + (tx + dx)] * c2w[ch * 9 + dy * 3 + dx];
  out[((size_t)b * nT + y0 + ty) * nT + x0 + tx] = acc;
}

// ---------------------------------------------------------------------------
// Kernel 2: projection GEMM  Y[M,512] = X[M,512] @ W[512,512]^T (+bias+add)
// fp32 inputs, bf16 output(s). One 16x16 tile per wave, 16 WMMAs (K=512).
// Per-lane fragment chunks are contiguous -> float4 loads + v_cvt_pk_bf16.
// ---------------------------------------------------------------------------
__global__ __launch_bounds__(256) void proj_bf16_kernel(
    const float* __restrict__ X, const float* __restrict__ W,
    const float* __restrict__ bias, const float* __restrict__ add0,
    const float* __restrict__ add1, __bf16* __restrict__ out0,
    __bf16* __restrict__ out1, int M) {
  const int wave = threadIdx.x >> 5;
  const int tile = blockIdx.x * 8 + wave;
  const int ntiles = (M >> 4) * (nD >> 4);
  if (tile >= ntiles) return;  // wave-uniform
  const int nt = tile & 31, mt = tile >> 5;
  const int m0 = mt << 4, n0 = nt << 4;
  const int lane = threadIdx.x & 31, q = lane & 15, hl = lane >> 4;

  v8f acc = {};
  const float* xrow = X + (size_t)(m0 + q) * nD;
  const float* wrow = W + (size_t)(n0 + q) * nD;
  for (int kt = 0; kt < 16; ++kt) {
    const int k0 = kt << 5;
    __builtin_prefetch(xrow + k0 + 64, 0, 0);  // next-next k-tile (speculative)
    __builtin_prefetch(wrow + k0 + 64, 0, 0);
    const float* xa = xrow + k0 + (hl << 4);    // A: one 16-float run
    float4x fa0 = *(const float4x*)(xa);
    float4x fa1 = *(const float4x*)(xa + 4);
    float4x fa2 = *(const float4x*)(xa + 8);
    float4x fa3 = *(const float4x*)(xa + 12);
    const float* wa = wrow + k0 + (hl << 4);    // B: one 16-float run
    float4x g0 = *(const float4x*)(wa);
    float4x g1 = *(const float4x*)(wa + 4);
    float4x g2 = *(const float4x*)(wa + 8);
    float4x g3 = *(const float4x*)(wa + 12);
    v16bf a, bw;
    pack8(a, 0, fa0, fa1); pack8(a, 8, fa2, fa3);
    pack8(bw, 0, g0, g1);  pack8(bw, 8, g2, g3);
    acc = wmma_bf16(a, bw, acc);
  }
  const int n = n0 + q;
  const float b0 = bias ? bias[n] : 0.f;
  const float e0 = b0 + (add0 ? add0[n] : 0.f);
#pragma unroll
  for (int i = 0; i < 8; ++i)
    out0[(size_t)(m0 + i + (hl << 3)) * nD + n] = (__bf16)(acc[i] + e0);
  if (out1) {
    const float e1 = b0 + (add1 ? add1[n] : 0.f);
#pragma unroll
    for (int i = 0; i < 8; ++i)
      out1[(size_t)(m0 + i + (hl << 3)) * nD + n] = (__bf16)(acc[i] + e1);
  }
}

// ---------------------------------------------------------------------------
// Kernel 3: fused attention. One wave per (b, h, 16-row t-tile).
// Relative shift folded: pos[t,s] = qv[t] . p[s-t+511]; only the 31-diagonal
// band c = s-t+15 is needed per 16x16 tile -> 2 WMMAs into a 16x32 LDS slab.
// Scores (16x512, pitch 516) stay in LDS; softmax in place; attn repacked as
// bf16; attn@V uses async-to-LDS staged V slabs (double buffered, pipelined).
// ---------------------------------------------------------------------------
__global__ __launch_bounds__(32) void attn_kernel(
    const __bf16* __restrict__ qu, const __bf16* __restrict__ qv,
    const __bf16* __restrict__ kmat, const __bf16* __restrict__ pmat,
    const __bf16* __restrict__ vmat, const float* __restrict__ bpp_bias,
    const unsigned char* __restrict__ mask, __bf16* __restrict__ ctx) {
  __shared__ __align__(16) float  slab[16 * 33];     // pos band tile
  __shared__ __align__(16) float  score[16 * 516];   // scores -> bf16 attn
  __shared__ __align__(16) __bf16 vslab[2 * 1024];   // 2 x (32x32) V k-slabs
  const int t0 = blockIdx.x << 4;
  const int h = blockIdx.y, b = blockIdx.z;
  const int lane = threadIdx.x, q = lane & 15, hl = lane >> 4;
  const size_t bh = ((size_t)b * nT) * nD + (size_t)h * nDH;

  // A fragments: (q+u_bias) and (q+v_bias) rows t0..t0+15, K = DH = 32.
  AFrag aqu, aqv;
  {
    const __bf16* r0 = qu + bh + (size_t)(t0 + q) * nD + (hl << 4);
    const __bf16* r1 = qv + bh + (size_t)(t0 + q) * nD + (hl << 4);
    aqu.q2[0] = *(const uint4x*)(r0);
    aqu.q2[1] = *(const uint4x*)(r0 + 8);
    aqv.q2[0] = *(const uint4x*)(r1);
    aqv.q2[1] = *(const uint4x*)(r1 + 8);
  }

  for (int st = 0; st < 32; ++st) {
    const int s0 = st << 4;
    const int rb = s0 - t0 + 496;  // p-row of band column c=0
    // --- pos band: PB[tt][c] = sum_d qv[t0+tt,d] * p[rb+c, d], c in [0,31]
#pragma unroll
    for (int pt = 0; pt < 2; ++pt) {
      AFrag bp;
      const int r = rb + (pt << 4) + q;
      if (r >= 0 && r < nP) {
        const __bf16* prow =
            pmat + ((size_t)b * nP + r) * nD + (size_t)h * nDH + (hl << 4);
        bp.q2[0] = *(const uint4x*)(prow);
        bp.q2[1] = *(const uint4x*)(prow + 8);
      } else {
        bp.q2[0] = (uint4x)0u;
        bp.q2[1] = (uint4x)0u;
      }
      v8f pacc = {};
      pacc = wmma_bf16(aqv.v, bp.v, pacc);
#pragma unroll
      for (int i = 0; i < 8; ++i)
        slab[(i + (hl << 3)) * 33 + (pt << 4) + q] = pacc[i];
    }
    // --- content: QK^T tile, single K=32 WMMA
    AFrag bk;
    {
      const __bf16* krow = kmat + bh + (size_t)(s0 + q) * nD + (hl << 4);
      bk.q2[0] = *(const uint4x*)(krow);
      bk.q2[1] = *(const uint4x*)(krow + 8);
    }
    v8f cacc = {};
    cacc = wmma_bf16(aqu.v, bk.v, cacc);
    // --- combine, scale, bpp bias, mask -> score LDS
    const int s = s0 + q;
    const bool masked = mask[b * nT + s] != 0;
#pragma unroll
    for (int i = 0; i < 8; ++i) {
      const int tt = i + (hl << 3);
      const int c = q - tt + 15;  // in [0,30] by construction
      float sc = (cacc[i] + slab[tt * 33 + c]) * 0.17677669529663687f
               + bpp_bias[((size_t)b * nT + t0 + tt) * nT + s];
      if (masked) sc = -10000.0f;
      score[tt * 516 + s] = sc;
    }
  }

  // --- softmax: one row per lane (lanes 0..15); re-pack attn as bf16 in place
  if (lane < 16) {
    float* row = &score[lane * 516];
    float mx = -3.0e38f;
    for (int c = 0; c < nT; ++c) mx = fmaxf(mx, row[c]);
    float sum = 0.f;
    for (int c = 0; c < nT; ++c) { float e = __expf(row[c] - mx); row[c] = e; sum += e; }
    const float inv = 1.0f / sum;
    unsigned short* rowh = (unsigned short*)row;  // bf16 write trails f32 read
    for (int c = 0; c < nT; ++c) {
      union { __bf16 b; unsigned short s; } cv;
      cv.b = (__bf16)(row[c] * inv);
      rowh[c] = cv.s;
    }
  }

  // --- ctx = attn @ V : V k-slabs async-staged into LDS, double buffered.
  const unsigned short* att = (const unsigned short*)score;  // pitch 1032 u16
  const __bf16* vs16 = (const __bf16*)vslab;
  const unsigned vbase = (unsigned)(unsigned long long)(size_t)vslab;  // LDS addr
  auto stage = [&](int kt) {
    const unsigned boff = vbase + ((unsigned)(kt & 1) << 11);
    const int row = lane >> 2, chunk = lane & 3;   // 8 rows x 4 chunks per op
#pragma unroll
    for (int op = 0; op < 4; ++op) {
      const int r = (op << 3) + row;
      const unsigned long long g = (unsigned long long)(size_t)(
          vmat + bh + (size_t)(((kt & 31) << 5) + r) * nD + (chunk << 3));
      const unsigned l = boff + (unsigned)(r * 64 + chunk * 16);
      asm volatile("global_load_async_to_lds_b128 %0, %1, off"
                   :: "v"(l), "v"(g) : "memory");
    }
  };
  stage(0);
  v8f acc0 = {}, acc1 = {};
  for (int kt = 0; kt < 32; ++kt) {
    stage(kt + 1);  // prefetch next slab into the other buffer
    // async loads complete in order: cnt<=4 retires the 4 older (current) ops
    asm volatile("s_wait_asynccnt 0x4" ::: "memory");
    const int k0 = kt << 5;
    AFrag a;  // attn A fragment, shared by both column tiles
    const unsigned short* arow = att + q * 1032 + k0 + (hl << 4);
    a.q2[0] = *(const uint4x*)(arow);
    a.q2[1] = *(const uint4x*)(arow + 8);
    const __bf16* vs = vs16 + ((kt & 1) << 10);
    v16bf b0, b1;
#pragma unroll
    for (int j = 0; j < 8; ++j) {
      const int kk = (hl << 4) + 2 * j;
      b0[2 * j]     = vs[kk * 32 + q];
      b0[2 * j + 1] = vs[(kk + 1) * 32 + q];
      b1[2 * j]     = vs[kk * 32 + 16 + q];
      b1[2 * j + 1] = vs[(kk + 1) * 32 + 16 + q];
    }
    acc0 = wmma_bf16(a.v, b0, acc0);
    acc1 = wmma_bf16(a.v, b1, acc1);
  }
#pragma unroll
  for (int i = 0; i < 8; ++i) {
    const size_t orow = bh + (size_t)(t0 + i + (hl << 3)) * nD + q;
    ctx[orow]      = (__bf16)acc0[i];
    ctx[orow + 16] = (__bf16)acc1[i];
  }
}

// ---------------------------------------------------------------------------
// Kernel 4: output projection  out[8192,512] = ctx_bf16 @ Wo^T + bo (fp32 out)
// ---------------------------------------------------------------------------
__global__ __launch_bounds__(256) void outproj_kernel(
    const __bf16* __restrict__ Xb, const float* __restrict__ W,
    const float* __restrict__ bias, float* __restrict__ out) {
  const int wave = threadIdx.x >> 5;
  const int tile = blockIdx.x * 8 + wave;
  const int nt = tile & 31, mt = tile >> 5;
  const int m0 = mt << 4, n0 = nt << 4;
  const int lane = threadIdx.x & 31, q = lane & 15, hl = lane >> 4;

  v8f acc = {};
  const __bf16* xrow = Xb + (size_t)(m0 + q) * nD;
  const float* wrow = W + (size_t)(n0 + q) * nD;
  for (int kt = 0; kt < 16; ++kt) {
    const int k0 = kt << 5;
    __builtin_prefetch(wrow + k0 + 64, 0, 0);
    AFrag a;
    const __bf16* xa = xrow + k0 + (hl << 4);
    a.q2[0] = *(const uint4x*)(xa);
    a.q2[1] = *(const uint4x*)(xa + 8);
    const float* wa = wrow + k0 + (hl << 4);
    float4x g0 = *(const float4x*)(wa);
    float4x g1 = *(const float4x*)(wa + 4);
    float4x g2 = *(const float4x*)(wa + 8);
    float4x g3 = *(const float4x*)(wa + 12);
    v16bf bw;
    pack8(bw, 0, g0, g1); pack8(bw, 8, g2, g3);
    acc = wmma_bf16(a.v, bw, acc);
  }
  const int n = n0 + q;
  const float bb = bias[n];
#pragma unroll
  for (int i = 0; i < 8; ++i)
    out[(size_t)(m0 + i + (hl << 3)) * nD + n] = acc[i] + bb;
}

// ---------------------------------------------------------------------------
extern "C" void kernel_launch(void* const* d_in, const int* in_sizes, int n_in,
                              void* d_out, int out_size, void* d_ws, size_t ws_size,
                              hipStream_t stream) {
  (void)in_sizes; (void)n_in; (void)out_size; (void)ws_size;
  const float* query = (const float*)d_in[0];
  const float* key_  = (const float*)d_in[1];
  const float* value = (const float*)d_in[2];
  const float* pose  = (const float*)d_in[3];
  const unsigned char* mask = (const unsigned char*)d_in[4];
  const float* bpp = (const float*)d_in[5];
  const float* Wq = (const float*)d_in[6];
  const float* bq = (const float*)d_in[7];
  const float* Wk = (const float*)d_in[8];
  const float* bk = (const float*)d_in[9];
  const float* Wv = (const float*)d_in[10];
  const float* bv = (const float*)d_in[11];
  const float* Wp = (const float*)d_in[12];
  const float* Wo = (const float*)d_in[13];
  const float* bo = (const float*)d_in[14];
  const float* ub = (const float*)d_in[15];   // u_bias[H,DH] == flat [512]
  const float* vb = (const float*)d_in[16];   // v_bias[H,DH] == flat [512]
  const float* c1w = (const float*)d_in[17];
  const float* c1b = (const float*)d_in[18];
  const float* c2w = (const float*)d_in[19];
  const float* c2b = (const float*)d_in[20];

  char* ws = (char*)d_ws;
  auto take = [&](size_t bytes) {
    char* p = ws;
    ws += (bytes + 255) & ~(size_t)255;
    return p;
  };
  float*  bpp_bias = (float*)take((size_t)nB * nT * nT * 4);   // 16.8 MB
  __bf16* quB = (__bf16*)take((size_t)nB * nT * nD * 2);       // 8.4 MB
  __bf16* qvB = (__bf16*)take((size_t)nB * nT * nD * 2);
  __bf16* kB  = (__bf16*)take((size_t)nB * nT * nD * 2);
  __bf16* vB  = (__bf16*)take((size_t)nB * nT * nD * 2);
  __bf16* pB  = (__bf16*)take((size_t)nB * nP * nD * 2);       // 16.8 MB
  __bf16* cB  = (__bf16*)take((size_t)nB * nT * nD * 2);

  conv_bias_kernel<<<dim3(nT / 16, nT / 16, nB), dim3(16, 16), 0, stream>>>(
      bpp, c1w, c1b, c2w, c2b, bpp_bias);

  // q projection writes both (q+u_bias) and (q+v_bias) bf16 copies
  proj_bf16_kernel<<<2048, 256, 0, stream>>>(query, Wq, bq, ub, vb, quB, qvB, nB * nT);
  proj_bf16_kernel<<<2048, 256, 0, stream>>>(key_,  Wk, bk, nullptr, nullptr, kB, nullptr, nB * nT);
  proj_bf16_kernel<<<2048, 256, 0, stream>>>(value, Wv, bv, nullptr, nullptr, vB, nullptr, nB * nT);
  proj_bf16_kernel<<<4092, 256, 0, stream>>>(pose,  Wp, nullptr, nullptr, nullptr, pB, nullptr, nB * nP);

  attn_kernel<<<dim3(nT / 16, nH, nB), 32, 0, stream>>>(
      quB, qvB, kB, pB, vB, bpp_bias, mask, cB);

  outproj_kernel<<<2048, 256, 0, stream>>>(cB, Wo, bo, (float*)d_out);
}
